// TraceFIM_28484223107681
// MI455X (gfx1250) — matrix operand order repaired and checked by reading, not compiled, not run on hardware
//
#include <hip/hip_runtime.h>
#include <hip/hip_bf16.h>

// TraceFIM for 2-layer MLP: B=32, D=3072, H=1024, C=100.
// Rank-1 trick: ||gW||^2 = ||input||^2 * ||dz||^2 per sample -> no per-sample
// weight-gradient materialization. All three matrix products (X@W1, h@W2,
// dlogits@W2^T) run on v_wmma_f32_16x16x32_f16 with in-register fp32->f16 cvt.
// W1 pass (12.6 MB) is K-split over 256 blocks to saturate HBM/L2; head GEMMs
// are spread over 28/64 blocks instead of one WGP. All reductions fixed-order.

#define D_DIM 3072
#define H_DIM 1024
#define C_DIM 100
#define C_PAD 112          // logits N padded to 7 n-tiles
#define K_PAD 128          // dlogits K padded for WMMA
#define B_DIM 32

#define KSPLIT   4
#define KCHUNK   (D_DIM / KSPLIT)           // 768
#define KWAVE    (KCHUNK / 4)               // 192 -> 6 wmma steps of 32

typedef __attribute__((ext_vector_type(16))) _Float16 v16h;
typedef __attribute__((ext_vector_type(8)))  float    v8f;

// workspace layout (floats)
#define WS_P1    0                                   // [4][32][1024]
#define WS_HBUF  (WS_P1 + KSPLIT * B_DIM * H_DIM)    // [32][1024]
#define WS_P2    (WS_HBUF + B_DIM * H_DIM)           // [4][32][112]
#define WS_DL    (WS_P2 + 4 * B_DIM * C_PAD)         // [32][128]
#define WS_DL2   (WS_DL + B_DIM * K_PAD)             // [32]
#define WS_H2    (WS_DL2 + B_DIM)                    // [32]
#define WS_X2    (WS_H2 + B_DIM)                     // [32]
#define WS_DZ2P  (WS_X2 + B_DIM)                     // [64][32]

// ---------------------------------------------------------------------------
// Kernel 1: partial Z1 tiles. Grid (64 n-tiles, 4 k-splits) x 8 waves
// (2 m-tiles x 4 k-subchunks). A/B fragments use contiguous k = 16*laneHi + j.
// ---------------------------------------------------------------------------
__global__ __launch_bounds__(256) void gemm1_partial_kernel(
    const float* __restrict__ X,    // [32, 3072]
    const float* __restrict__ W1,   // [3072, 1024]
    float* __restrict__ P)          // [4][32][1024] partials
{
    __shared__ float red[8][32][8];

    const int wave   = threadIdx.x >> 5;
    const int lane   = threadIdx.x & 31;
    const int laneLo = lane & 15;
    const int laneHi = lane >> 4;

    const int mtile = wave & 1;
    const int ksub  = wave >> 1;
    const int m0    = mtile * 16;
    const int n0    = blockIdx.x * 16;
    const int kbase = blockIdx.y * KCHUNK + ksub * KWAVE;

    const float* __restrict__ xrow = X  + (m0 + laneLo) * D_DIM;
    const float* __restrict__ wcol = W1 + (n0 + laneLo);

    v8f acc = {};
    for (int k0 = kbase; k0 < kbase + KWAVE; k0 += 32) {
        // branchless speculative prefetch of next K-step (global_prefetch_b8)
        const int kpf = (k0 + 32 < D_DIM) ? (k0 + 32) : k0;
        __builtin_prefetch(wcol + kpf * H_DIM, 0, 1);

        v16h a, b;
        const int kaBase = k0 + laneHi * 16;
#pragma unroll
        for (int j = 0; j < 8; ++j) {
            a[j]     = (_Float16)xrow[kaBase + j];
            a[j + 8] = (_Float16)xrow[kaBase + 8 + j];
        }
        const int kbBase = k0 + laneHi * 16;
#pragma unroll
        for (int j = 0; j < 16; ++j)
            b[j] = (_Float16)wcol[(kbBase + j) * H_DIM];

        acc = __builtin_amdgcn_wmma_f32_16x16x32_f16(
            false, a, false, b, (short)0, acc, false, false);
    }

#pragma unroll
    for (int r = 0; r < 8; ++r) red[wave][lane][r] = acc[r];
    __syncthreads();

    if (wave < 2) {
        const int mt = wave;
        float* __restrict__ Pp = P + blockIdx.y * (B_DIM * H_DIM);
        const int n = n0 + laneLo;
#pragma unroll
        for (int r = 0; r < 8; ++r) {
            float s = red[0 + mt][lane][r] + red[2 + mt][lane][r]
                    + red[4 + mt][lane][r] + red[6 + mt][lane][r];
            const int m = mt * 16 + laneHi * 8 + r;
            Pp[m * H_DIM + n] = s;
        }
    }
}

// ---------------------------------------------------------------------------
// Kernel 2: h = relu(sum_k P1[k] + b1), fixed-order (deterministic).
// ---------------------------------------------------------------------------
__global__ __launch_bounds__(256) void combine_relu_kernel(
    const float* __restrict__ P, const float* __restrict__ b1,
    float* __restrict__ Hout)
{
    const int idx = blockIdx.x * blockDim.x + threadIdx.x;   // 0..32767
    const int n = idx & (H_DIM - 1);
    float z = b1[n];
#pragma unroll
    for (int k = 0; k < KSPLIT; ++k) z += P[k * (B_DIM * H_DIM) + idx];
    Hout[idx] = z > 0.f ? z : 0.f;
}

// ---------------------------------------------------------------------------
// Kernel 3: logits partials = h @ W2 (N padded to 112, zero-guarded columns).
// Grid (7 n-tiles, 4 k-splits) x 8 waves (2 m x 4 ksub; K=1024 -> 2 steps).
// ---------------------------------------------------------------------------
__global__ __launch_bounds__(256) void gemm2_partial_kernel(
    const float* __restrict__ Hbuf,  // [32, 1024]
    const float* __restrict__ W2,    // [1024, 100]
    float* __restrict__ P2)          // [4][32][112]
{
    __shared__ float red[8][32][8];

    const int wave   = threadIdx.x >> 5;
    const int lane   = threadIdx.x & 31;
    const int laneLo = lane & 15;
    const int laneHi = lane >> 4;

    const int mtile = wave & 1;
    const int ksub  = wave >> 1;
    const int m0    = mtile * 16;
    const int n0    = blockIdx.x * 16;
    const int kbase = blockIdx.y * 256 + ksub * 64;

    const float* __restrict__ hrow = Hbuf + (m0 + laneLo) * H_DIM;
    const int n = n0 + laneLo;
    const bool nok = (n < C_DIM);

    v8f acc = {};
    for (int k0 = kbase; k0 < kbase + 64; k0 += 32) {
        v16h a, b;
        const int kaBase = k0 + laneHi * 16;
#pragma unroll
        for (int j = 0; j < 8; ++j) {
            a[j]     = (_Float16)hrow[kaBase + j];
            a[j + 8] = (_Float16)hrow[kaBase + 8 + j];
        }
        const int kbBase = k0 + laneHi * 16;
#pragma unroll
        for (int j = 0; j < 16; ++j)
            b[j] = nok ? (_Float16)W2[(kbBase + j) * C_DIM + n] : (_Float16)0.f;

        acc = __builtin_amdgcn_wmma_f32_16x16x32_f16(
            false, a, false, b, (short)0, acc, false, false);
    }

#pragma unroll
    for (int r = 0; r < 8; ++r) red[wave][lane][r] = acc[r];
    __syncthreads();

    if (wave < 2) {
        const int mt = wave;
        float* __restrict__ Pp = P2 + blockIdx.y * (B_DIM * C_PAD);
#pragma unroll
        for (int r = 0; r < 8; ++r) {
            float s = red[0 + mt][lane][r] + red[2 + mt][lane][r]
                    + red[4 + mt][lane][r] + red[6 + mt][lane][r];
            const int m = mt * 16 + laneHi * 8 + r;
            Pp[m * C_PAD + n] = s;
        }
    }
}

// ---------------------------------------------------------------------------
// Kernel 4: softmax + categorical sample + dlogits + per-sample norms.
// One block, 32 waves (one per sample); only ~100 exps/sample -> cheap.
// ---------------------------------------------------------------------------
__global__ __launch_bounds__(1024) void softmax_sample_kernel(
    const float* __restrict__ X,     // [32, 3072]
    const float* __restrict__ Hbuf,  // [32, 1024]
    const float* __restrict__ P2,    // [4][32][112]
    const float* __restrict__ b2,    // [100]
    float* __restrict__ dlbuf,       // [32][128], zero-padded
    float* __restrict__ dl2buf,      // [32]
    float* __restrict__ h2buf,       // [32]
    float* __restrict__ x2buf)       // [32]
{
    __shared__ float s_p[B_DIM][C_PAD];

    const int wave = threadIdx.x >> 5;   // sample
    const int lane = threadIdx.x & 31;

    float logit[4];
#pragma unroll
    for (int t = 0; t < 4; ++t) {
        const int c = lane + 32 * t;
        float a = -1e30f;
        if (c < C_DIM) {
            a = b2[c];
#pragma unroll
            for (int ks = 0; ks < 4; ++ks)
                a += P2[ks * (B_DIM * C_PAD) + wave * C_PAD + c];
        }
        logit[t] = a;
    }

    float mx = fmaxf(fmaxf(logit[0], logit[1]), fmaxf(logit[2], logit[3]));
#pragma unroll
    for (int off = 16; off >= 1; off >>= 1) mx = fmaxf(mx, __shfl_xor(mx, off, 32));
    float e[4], sum = 0.f;
#pragma unroll
    for (int t = 0; t < 4; ++t) {
        const int c = lane + 32 * t;
        e[t] = (c < C_DIM) ? __expf(logit[t] - mx) : 0.f;
        sum += e[t];
    }
#pragma unroll
    for (int off = 16; off >= 1; off >>= 1) sum += __shfl_xor(sum, off, 32);
    const float inv = 1.f / sum;
#pragma unroll
    for (int t = 0; t < 4; ++t) {
        const int c = lane + 32 * t;
        if (c < C_DIM) s_p[wave][c] = e[t] * inv;
    }
    __syncthreads();

    // y ~ Categorical(p): hash-based stand-in for jax threefry
    int y = 0;
    if (lane == 0) {
        unsigned s = 0x9E3779B9u * (unsigned)(wave + 1) ^ 0x85EBCA6Bu;
        s ^= s >> 16; s *= 0x7FEB352Du; s ^= s >> 15; s *= 0x846CA68Bu; s ^= s >> 16;
        const float u = (float)(s >> 8) * (1.0f / 16777216.0f);
        float csum = 0.f; y = C_DIM - 1;
        for (int c = 0; c < C_DIM; ++c) {
            csum += s_p[wave][c];
            if (u < csum) { y = c; break; }
        }
    }
    y = __shfl(y, 0, 32);

    float dl2 = 0.f;
#pragma unroll
    for (int t = 0; t < 4; ++t) {
        const int c = lane + 32 * t;       // covers 0..127 exactly
        float dl = 0.f;
        if (c < C_DIM) dl = e[t] * inv - (c == y ? 1.f : 0.f);
        dlbuf[wave * K_PAD + c] = dl;      // zero padding for c >= 100
        dl2 += dl * dl;
    }

    float h2 = 0.f;
    for (int k = lane; k < H_DIM; k += 32) { const float v = Hbuf[wave * H_DIM + k]; h2 += v * v; }
    float x2 = 0.f;
    for (int d = lane; d < D_DIM; d += 32) { const float v = X[wave * D_DIM + d]; x2 += v * v; }
#pragma unroll
    for (int off = 16; off >= 1; off >>= 1) {
        dl2 += __shfl_xor(dl2, off, 32);
        h2  += __shfl_xor(h2,  off, 32);
        x2  += __shfl_xor(x2,  off, 32);
    }
    if (lane == 0) { dl2buf[wave] = dl2; h2buf[wave] = h2; x2buf[wave] = x2; }
}

// ---------------------------------------------------------------------------
// Kernel 5: dh = dlogits @ W2^T (K padded to 128), dz = dh * 1[h>0], and
// per-sample partial ||dz||^2 for this 16-column slab. Grid 64 blocks x
// 8 waves (2 m x 4 ksub, 1 wmma step each). B reads W2[n*100+c]: contiguous.
// ---------------------------------------------------------------------------
__global__ __launch_bounds__(256) void gemm3_dz_kernel(
    const float* __restrict__ dlbuf, // [32][128]
    const float* __restrict__ W2,    // [1024, 100]
    const float* __restrict__ Hbuf,  // [32, 1024]
    float* __restrict__ dz2p)        // [64][32]
{
    __shared__ float red[8][32][8];

    const int wave   = threadIdx.x >> 5;
    const int lane   = threadIdx.x & 31;
    const int laneLo = lane & 15;
    const int laneHi = lane >> 4;

    const int mtile = wave & 1;
    const int ksub  = wave >> 1;
    const int m0    = mtile * 16;
    const int n0    = blockIdx.x * 16;
    const int kbase = ksub * 32;

    const float* __restrict__ dlrow = dlbuf + (m0 + laneLo) * K_PAD;
    const int n = n0 + laneLo;

    v16h a, b;
    const int kaBase = kbase + laneHi * 16;
#pragma unroll
    for (int j = 0; j < 8; ++j) {
        a[j]     = (_Float16)dlrow[kaBase + j];
        a[j + 8] = (_Float16)dlrow[kaBase + 8 + j];
    }
    const int kbBase = kbase + laneHi * 16;
#pragma unroll
    for (int j = 0; j < 16; ++j) {
        const int c = kbBase + j;          // k index = class index
        b[j] = (c < C_DIM) ? (_Float16)W2[n * C_DIM + c] : (_Float16)0.f;
    }
    v8f acc = {};
    acc = __builtin_amdgcn_wmma_f32_16x16x32_f16(
        false, a, false, b, (short)0, acc, false, false);

#pragma unroll
    for (int r = 0; r < 8; ++r) red[wave][lane][r] = acc[r];
    __syncthreads();

    if (wave < 2) {
        const int mt = wave;
#pragma unroll
        for (int r = 0; r < 8; ++r) {
            float dh = red[0 + mt][lane][r] + red[2 + mt][lane][r]
                     + red[4 + mt][lane][r] + red[6 + mt][lane][r];
            const int m = mt * 16 + laneHi * 8 + r;
            const float hk = Hbuf[m * H_DIM + n];
            float s = (hk > 0.f) ? dh * dh : 0.f;
            // sum over the 16 columns held by this 16-lane half
            s += __shfl_xor(s, 1, 32);
            s += __shfl_xor(s, 2, 32);
            s += __shfl_xor(s, 4, 32);
            s += __shfl_xor(s, 8, 32);
            if (laneLo == 0) dz2p[blockIdx.x * B_DIM + m] = s;
        }
    }
}

// ---------------------------------------------------------------------------
// Kernel 6: final fixed-order reduction -> 5 outputs.
// ---------------------------------------------------------------------------
__global__ __launch_bounds__(1024) void finalize_kernel(
    const float* __restrict__ dz2p,  // [64][32]
    const float* __restrict__ x2buf, const float* __restrict__ h2buf,
    const float* __restrict__ dl2buf, float* __restrict__ out)
{
    __shared__ float s_acc[B_DIM][4];

    const int wave = threadIdx.x >> 5;   // sample
    const int lane = threadIdx.x & 31;

    float s = dz2p[lane * B_DIM + wave] + dz2p[(lane + 32) * B_DIM + wave];
#pragma unroll
    for (int off = 16; off >= 1; off >>= 1) s += __shfl_xor(s, off, 32);

    if (lane == 0) {
        const float dl2 = dl2buf[wave];
        s_acc[wave][0] = x2buf[wave] * s;   // trace(gW1)
        s_acc[wave][1] = s;                 // trace(gb1)
        s_acc[wave][2] = h2buf[wave] * dl2; // trace(gW2)
        s_acc[wave][3] = dl2;               // trace(gb2)
    }
    __syncthreads();

    if (wave == 0) {
        float v0 = s_acc[lane][0], v1 = s_acc[lane][1];
        float v2 = s_acc[lane][2], v3 = s_acc[lane][3];
#pragma unroll
        for (int off = 16; off >= 1; off >>= 1) {
            v0 += __shfl_xor(v0, off, 32);
            v1 += __shfl_xor(v1, off, 32);
            v2 += __shfl_xor(v2, off, 32);
            v3 += __shfl_xor(v3, off, 32);
        }
        if (lane == 0) {
            const float invB = 1.0f / (float)B_DIM;
            const float t0 = v0 * invB, t1 = v1 * invB;
            const float t2 = v2 * invB, t3 = v3 * invB;
            out[0] = t0; out[1] = t1; out[2] = t2; out[3] = t3;
            out[4] = t0 + t1 + t2 + t3;
        }
    }
}

extern "C" void kernel_launch(void* const* d_in, const int* in_sizes, int n_in,
                              void* d_out, int out_size, void* d_ws, size_t ws_size,
                              hipStream_t stream) {
    (void)in_sizes; (void)n_in; (void)out_size; (void)ws_size;
    const float* x  = (const float*)d_in[0];
    const float* W1 = (const float*)d_in[1];
    const float* b1 = (const float*)d_in[2];
    const float* W2 = (const float*)d_in[3];
    const float* b2 = (const float*)d_in[4];
    float* out = (float*)d_out;

    float* ws    = (float*)d_ws;
    float* P1    = ws + WS_P1;
    float* hbuf  = ws + WS_HBUF;
    float* P2    = ws + WS_P2;
    float* dlbuf = ws + WS_DL;
    float* dl2b  = ws + WS_DL2;
    float* h2b   = ws + WS_H2;
    float* x2b   = ws + WS_X2;
    float* dz2p  = ws + WS_DZ2P;

    gemm1_partial_kernel<<<dim3(H_DIM / 16, KSPLIT), 256, 0, stream>>>(x, W1, P1);
    combine_relu_kernel<<<(B_DIM * H_DIM) / 256, 256, 0, stream>>>(P1, b1, hbuf);
    gemm2_partial_kernel<<<dim3(C_PAD / 16, 4), 256, 0, stream>>>(hbuf, W2, P2);
    softmax_sample_kernel<<<1, 1024, 0, stream>>>(x, hbuf, P2, b2,
                                                  dlbuf, dl2b, h2b, x2b);
    gemm3_dz_kernel<<<H_DIM / 16, 256, 0, stream>>>(dlbuf, W2, hbuf, dz2p);
    finalize_kernel<<<1, 1024, 0, stream>>>(dz2p, x2b, h2b, dl2b, out);
}
